// AttentionMechanism_28587302323079
// MI455X (gfx1250) — compile-verified, matrix operand order checked by reading, not run on hardware
//
#include <hip/hip_runtime.h>
#include <math.h>

#define B_   128
#define C_   2048
#define S_   196
#define SP   208      // spatial padded to 13*16
#define T_   20
#define HID_ 1024

typedef __attribute__((ext_vector_type(16))) __bf16 v16bf;
typedef __attribute__((ext_vector_type(8)))  float  v8f;

union Frag {
    v16bf v;
    uint4 q[2];
};

__device__ __forceinline__ unsigned short f2bf(float x) {
    __bf16 h = (__bf16)x;
    return __builtin_bit_cast(unsigned short, h);
}
__device__ __forceinline__ float bf2f(unsigned short u) {
    __bf16 h = __builtin_bit_cast(__bf16, u);
    return (float)h;
}

// tanh(x) = sign(x)*(1 - 2/(exp(2|x|)+1)) with native exp + native rcp:
// one v_exp_f32 + one v_rcp_f32 + a few VALU. exp overflow -> rcp(inf)=0 -> 1.
__device__ __forceinline__ float fast_tanh(float x) {
    float ax = fabsf(x);
    float e  = __expf(2.0f * ax);
    float t  = 1.0f - 2.0f * __builtin_amdgcn_rcpf(e + 1.0f);
    return copysignf(t, x);
}

// ---------------------------------------------------------------- utilities

__global__ __launch_bounds__(256) void mean_kernel(const float* __restrict__ vq,
                                                   float* __restrict__ u) {
    int idx = blockIdx.x * 256 + threadIdx.x;       // B*HID threads
    int b = idx >> 10, h = idx & 1023;
    float s = 0.f;
    #pragma unroll
    for (int t = 0; t < T_; ++t)
        s += vq[((size_t)b * T_ + t) * HID_ + h];
    u[idx] = s * (1.0f / T_);
}

__global__ __launch_bounds__(256) void cvt_bf16_kernel(const float* __restrict__ in,
                                                       unsigned short* __restrict__ out,
                                                       int n) {
    int i = blockIdx.x * 256 + threadIdx.x;
    if (i < n) out[i] = f2bf(in[i]);
}

// v_q_t[b,n] = b_u[n] + sum_h u[b,h] * w_u[n,h]
__global__ __launch_bounds__(256) void qt_kernel(const float* __restrict__ u,
                                                 const float* __restrict__ w_u,
                                                 const float* __restrict__ b_u,
                                                 float* __restrict__ out) {
    int idx = blockIdx.x * 256 + threadIdx.x;       // B*HID threads
    int b = idx >> 10, n = idx & 1023;
    const float4* ur = (const float4*)(u + (size_t)b * HID_);
    const float4* wr = (const float4*)(w_u + (size_t)n * HID_);
    float s = b_u[n];
    #pragma unroll 4
    for (int h = 0; h < HID_ / 4; ++h) {
        float4 a = ur[h], w = wr[h];
        s += a.x * w.x + a.y * w.y + a.z * w.z + a.w * w.w;
    }
    out[idx] = s;
}

// ---------------------------------------------------------------- GEMM1
// vi[b,s,n] = tanh( sum_c v_i[b,c,s] * l1_w[n,c] + l1_b[n] ), stored bf16 [B,SP,HID]
// grid: (16 n-blocks, 128 batches), 256 threads = 8 waves.
// A operand transposed+converted through double-buffered LDS in WMMA A-frag layout.
__global__ __launch_bounds__(256) void gemm1_tanh_kernel(
        const float* __restrict__ vI,             // [B, C, S] f32
        const unsigned short* __restrict__ Wbf,   // l1_w bf16 [HID, C]
        const float* __restrict__ bias,           // [HID]
        unsigned short* __restrict__ viOut)       // [B, SP, HID] bf16
{
    const int b    = blockIdx.y;
    const int nb   = blockIdx.x * 64;
    const int tid  = threadIdx.x;
    const int wave = tid >> 5;
    const int lane = tid & 31;
    const int lg   = lane >> 4;      // lane group (0: K 0-7/16-23, 1: K 8-15/24-31)
    const int lm   = lane & 15;

    __shared__ unsigned short ldsA[2][13 * 32 * 16];   // double-buffered frag-layout bf16

    const int t1 = wave;             // m-tile 1 (0..7)
    const int t2 = wave + 8;         // m-tile 2 (8..15; valid if <13)
    // force a scalar branch: keep EXEC all-ones around WMMA
    const bool has2 = __builtin_amdgcn_readfirstlane((int)(t2 < 13)) != 0;

    v8f acc1[4], acc2[4];
    const v8f vzero = {0.f, 0.f, 0.f, 0.f, 0.f, 0.f, 0.f, 0.f};
    #pragma unroll
    for (int i = 0; i < 4; ++i) { acc1[i] = vzero; acc2[i] = vzero; }

    const float* vIb = vI + (size_t)b * C_ * S_;

    // staging decomposition without div/mod: thread owns kk = tid>>3,
    // and 26 consecutive s starting at (tid&7)*26. 8 threads cover one kk row.
    const int st_kk = tid >> 3;           // 0..31
    const int st_s0 = (tid & 7) * 26;     // 0,26,...,182
    const int st_g   = (st_kk >> 3) & 1;
    const int st_pos = (st_kk & 7) + ((st_kk >> 4) << 3);

    auto stageA = [&](int ks, int buf) {
        const float* src = vIb + (size_t)(ks * 32 + st_kk) * S_;
        unsigned short* dst = &ldsA[buf][0];
        #pragma unroll 2
        for (int i = 0; i < 26; ++i) {
            int s = st_s0 + i;
            float val = (s < S_) ? src[s] : 0.0f;
            int t = s >> 4;
            int m = s & 15;
            dst[((t * 32) + st_g * 16 + m) * 16 + st_pos] = f2bf(val);
        }
    };

    // hoisted per-lane row pointers for B frags
    const unsigned short* wrow[4];
    #pragma unroll
    for (int nf = 0; nf < 4; ++nf)
        wrow[nf] = Wbf + (size_t)(nb + nf * 16 + lm) * C_ + lg * 8;

    const int NKS = C_ / 32;                      // 64 K-steps
    stageA(0, 0);

    for (int ks = 0; ks < NKS; ++ks) {
        __syncthreads();                          // buffer `cur` staged; prev reads done
        const int cur = ks & 1;
        if (ks + 1 < NKS) stageA(ks + 1, cur ^ 1);   // overlap next stage with compute

        const int k0 = ks * 32;

        Frag bf[4];
        #pragma unroll
        for (int nf = 0; nf < 4; ++nf) {
            bf[nf].q[0] = *(const uint4*)(wrow[nf] + k0);
            bf[nf].q[1] = *(const uint4*)(wrow[nf] + k0 + 16);
        }

        Frag a1;
        a1.q[0] = *(const uint4*)&ldsA[cur][(t1 * 32 + lane) * 16];
        a1.q[1] = *(const uint4*)&ldsA[cur][(t1 * 32 + lane) * 16 + 8];
        #pragma unroll
        for (int nf = 0; nf < 4; ++nf)
            acc1[nf] = __builtin_amdgcn_wmma_f32_16x16x32_bf16(
                false, a1.v, false, bf[nf].v, (short)0, acc1[nf], false, false);

        if (has2) {
            Frag a2;
            a2.q[0] = *(const uint4*)&ldsA[cur][(t2 * 32 + lane) * 16];
            a2.q[1] = *(const uint4*)&ldsA[cur][(t2 * 32 + lane) * 16 + 8];
            #pragma unroll
            for (int nf = 0; nf < 4; ++nf)
                acc2[nf] = __builtin_amdgcn_wmma_f32_16x16x32_bf16(
                    false, a2.v, false, bf[nf].v, (short)0, acc2[nf], false, false);
        }
    }

    // ---- epilogue: +bias, tanh, store bf16
    unsigned short* outB = viOut + (size_t)b * SP * HID_;
    #pragma unroll
    for (int nf = 0; nf < 4; ++nf) {
        int n = nb + nf * 16 + lm;
        float bv = bias[n];
        #pragma unroll
        for (int r = 0; r < 8; ++r) {
            int m = r + 8 * lg;
            int s1 = t1 * 16 + m;
            outB[(size_t)s1 * HID_ + n] = f2bf(fast_tanh(acc1[nf][r] + bv));
            if (has2) {
                int s2 = t2 * 16 + m;
                outB[(size_t)s2 * HID_ + n] = f2bf(fast_tanh(acc2[nf][r] + bv));
            }
        }
    }
}

// ---------------------------------------------------------------- fused hop
// For its (b, 64-col block): H[s,n] = tanh( vi[b,s,:] . w_vi[n,:] + qt[b,n] ),
// softmax over s, u_out[b,n] = u_in[b,n] + sum_s p[s,n]*vi[b,s,n].
__global__ __launch_bounds__(256) void attn_hop_kernel(
        const unsigned short* __restrict__ viBf,  // [B, SP, HID] bf16
        const unsigned short* __restrict__ Wbf,   // w_vi bf16 [HID, HID]
        const float* __restrict__ qt,             // [B, HID]
        const float* __restrict__ uIn,            // [B, HID]
        float* __restrict__ uOut)                 // [B, HID]
{
    const int b    = blockIdx.y;
    const int nb   = blockIdx.x * 64;
    const int tid  = threadIdx.x;
    const int wave = tid >> 5;
    const int lane = tid & 31;
    const int lg   = lane >> 4;
    const int lm   = lane & 15;

    __shared__ float ha[SP * 64];      // 53 KB strip (of 320 KB WGP LDS)
    __shared__ float smax[256];
    __shared__ float ssum[256];
    __shared__ float sdot[256];
    __shared__ float colmax[64];

    const unsigned short* viB = viBf + (size_t)b * SP * HID_;

    const int t1 = wave;
    const int t2 = wave + 8;
    const bool has2 = __builtin_amdgcn_readfirstlane((int)(t2 < 13)) != 0;

    v8f acc1[4], acc2[4];
    const v8f vzero = {0.f, 0.f, 0.f, 0.f, 0.f, 0.f, 0.f, 0.f};
    #pragma unroll
    for (int i = 0; i < 4; ++i) { acc1[i] = vzero; acc2[i] = vzero; }

    // hoisted per-lane row pointers: K offsets become instruction immediates
    const unsigned short* wrow[4];
    #pragma unroll
    for (int nf = 0; nf < 4; ++nf)
        wrow[nf] = Wbf + (size_t)(nb + nf * 16 + lm) * HID_ + lg * 8;
    const unsigned short* arow1 = viB + (size_t)(t1 * 16 + lm) * HID_ + lg * 8;
    const unsigned short* arow2 = viB + (size_t)(t2 * 16 + lm) * HID_ + lg * 8;

    #pragma unroll 4
    for (int k0 = 0; k0 < HID_; k0 += 32) {
        Frag bf[4];
        #pragma unroll
        for (int nf = 0; nf < 4; ++nf) {
            bf[nf].q[0] = *(const uint4*)(wrow[nf] + k0);
            bf[nf].q[1] = *(const uint4*)(wrow[nf] + k0 + 16);
            if (k0 + 128 < HID_)
                __builtin_prefetch(wrow[nf] + k0 + 128, 0, 3);   // global_prefetch ahead
        }
        Frag a1;
        a1.q[0] = *(const uint4*)(arow1 + k0);
        a1.q[1] = *(const uint4*)(arow1 + k0 + 16);
        #pragma unroll
        for (int nf = 0; nf < 4; ++nf)
            acc1[nf] = __builtin_amdgcn_wmma_f32_16x16x32_bf16(
                false, a1.v, false, bf[nf].v, (short)0, acc1[nf], false, false);

        if (has2) {
            Frag a2;
            a2.q[0] = *(const uint4*)(arow2 + k0);
            a2.q[1] = *(const uint4*)(arow2 + k0 + 16);
            #pragma unroll
            for (int nf = 0; nf < 4; ++nf)
                acc2[nf] = __builtin_amdgcn_wmma_f32_16x16x32_bf16(
                    false, a2.v, false, bf[nf].v, (short)0, acc2[nf], false, false);
        }
    }

    // ---- epilogue into LDS strip: h_a = tanh(acc + qt); pad rows -> -1e30
    #pragma unroll
    for (int nf = 0; nf < 4; ++nf) {
        int nc = nf * 16 + lm;                     // column within block
        float q = qt[(size_t)b * HID_ + nb + nc];
        #pragma unroll
        for (int r = 0; r < 8; ++r) {
            int m = r + 8 * lg;
            int s1 = t1 * 16 + m;
            ha[s1 * 64 + nc] = fast_tanh(acc1[nf][r] + q);   // t1 rows always < 128 < S_
            if (has2) {
                int s2 = t2 * 16 + m;
                float v = fast_tanh(acc2[nf][r] + q);
                ha[s2 * 64 + nc] = (s2 < S_) ? v : -1.0e30f; // pad rows killed here
            }
        }
    }
    __syncthreads();

    // ---- softmax over s + weighted sum with vi, per column (branch-free loops)
    const int n     = tid & 63;       // column 0..63
    const int chunk = tid >> 6;       // 0..3, 52 rows each
    const int s0 = chunk * 52;

    float mx = -3.0e38f;
    #pragma unroll 4
    for (int i = 0; i < 52; ++i)
        mx = fmaxf(mx, ha[(s0 + i) * 64 + n]);
    smax[tid] = mx;
    __syncthreads();
    if (tid < 64)
        colmax[tid] = fmaxf(fmaxf(smax[tid], smax[tid + 64]),
                            fmaxf(smax[tid + 128], smax[tid + 192]));
    __syncthreads();

    float cm = colmax[n];
    float sum = 0.f, dot = 0.f;
    #pragma unroll 4
    for (int i = 0; i < 52; ++i) {
        int s = s0 + i;
        float e = __expf(ha[s * 64 + n] - cm);     // pad rows: exp(-huge) = 0
        sum += e;
        dot += e * bf2f(viB[(size_t)s * HID_ + nb + n]);
    }
    ssum[tid] = sum;
    sdot[tid] = dot;
    __syncthreads();

    if (tid < 64) {
        float tot  = ssum[tid] + ssum[tid + 64] + ssum[tid + 128] + ssum[tid + 192];
        float dtot = sdot[tid] + sdot[tid + 64] + sdot[tid + 128] + sdot[tid + 192];
        size_t oi = (size_t)b * HID_ + nb + tid;
        uOut[oi] = uIn[oi] + dtot / tot;
    }
}

// ---------------------------------------------------------------- launch

extern "C" void kernel_launch(void* const* d_in, const int* in_sizes, int n_in,
                              void* d_out, int out_size, void* d_ws, size_t ws_size,
                              hipStream_t stream) {
    const float* v_i  = (const float*)d_in[0];   // [B,C,H,W]
    const float* v_q  = (const float*)d_in[1];   // [B,T,HID]
    const float* l1_w = (const float*)d_in[2];   // [HID,C]
    const float* l1_b = (const float*)d_in[3];   // [HID]
    const float* w_vi0 = (const float*)d_in[4];  // [HID,HID]
    const float* w_u0  = (const float*)d_in[5];
    const float* b_u0  = (const float*)d_in[6];
    const float* w_vi1 = (const float*)d_in[7];
    const float* w_u1  = (const float*)d_in[8];
    const float* b_u1  = (const float*)d_in[9];
    float* out = (float*)d_out;                  // [B,HID]

    // workspace layout
    char* p = (char*)d_ws;
    unsigned short* vi_bf   = (unsigned short*)p; p += (size_t)B_ * SP * HID_ * 2;   // 54.5 MB
    unsigned short* l1w_bf  = (unsigned short*)p; p += (size_t)HID_ * C_ * 2;        // 4 MB
    unsigned short* wvi0_bf = (unsigned short*)p; p += (size_t)HID_ * HID_ * 2;      // 2 MB
    unsigned short* wvi1_bf = (unsigned short*)p; p += (size_t)HID_ * HID_ * 2;      // 2 MB
    float* u_a = (float*)p; p += (size_t)B_ * HID_ * 4;
    float* u_b = (float*)p; p += (size_t)B_ * HID_ * 4;
    float* qtv = (float*)p; p += (size_t)B_ * HID_ * 4;

    const int BH = B_ * HID_;                    // 131072

    // 1) u0 = mean_t v_q
    mean_kernel<<<BH / 256, 256, 0, stream>>>(v_q, u_a);

    // 2) weight conversions to bf16
    cvt_bf16_kernel<<<(HID_ * C_) / 256, 256, 0, stream>>>(l1_w, l1w_bf, HID_ * C_);
    cvt_bf16_kernel<<<(HID_ * HID_) / 256, 256, 0, stream>>>(w_vi0, wvi0_bf, HID_ * HID_);
    cvt_bf16_kernel<<<(HID_ * HID_) / 256, 256, 0, stream>>>(w_vi1, wvi1_bf, HID_ * HID_);

    // 3) GEMM1 + tanh -> vi (bf16, padded S)
    {
        dim3 grid(HID_ / 64, B_);
        gemm1_tanh_kernel<<<grid, 256, 0, stream>>>(v_i, l1w_bf, l1_b, vi_bf);
    }

    // 4) hop 0
    qt_kernel<<<BH / 256, 256, 0, stream>>>(u_a, w_u0, b_u0, qtv);
    {
        dim3 grid(HID_ / 64, B_);
        attn_hop_kernel<<<grid, 256, 0, stream>>>(vi_bf, wvi0_bf, qtv, u_a, u_b);
    }

    // 5) hop 1 -> writes d_out directly
    qt_kernel<<<BH / 256, 256, 0, stream>>>(u_b, w_u1, b_u1, qtv);
    {
        dim3 grid(HID_ / 64, B_);
        attn_hop_kernel<<<grid, 256, 0, stream>>>(vi_bf, wvi1_bf, qtv, u_b, out);
    }
    (void)in_sizes; (void)n_in; (void)out_size; (void)ws_size;
}
